// ESNModel_11682311045218
// MI455X (gfx1250) — matrix operand order repaired and checked by reading, not compiled, not run on hardware
//
#include <hip/hip_runtime.h>
#include <cstdint>
#include <cstddef>

// ---------------------------------------------------------------------------
// ESN forward, MI455X (gfx1250).
//   R=8192, D=64, 1000 warm steps + 1000 free-running prediction steps.
//   Dominant cost: W_h@h each step. W_h converted to bf16 (128MB -> fits 192MB
//   L2) and pre-packed into the v_wmma_f32_16x16x32_bf16 A-operand lane layout.
//   One persistent kernel runs all 2000 steps with device-wide atomic barriers.
//   Inner loop uses two independent WMMA accumulator chains to avoid XDL
//   TRANS32_DEP stalls between back-to-back dependent WMMAs.
// ---------------------------------------------------------------------------

typedef __bf16  v16bf __attribute__((ext_vector_type(16)));
typedef float   v8f   __attribute__((ext_vector_type(8)));

#define R        8192
#define DIM      64
#define DYN      1000
#define STEPS    2000
#define NWG      256        // persistent workgroups (all must be co-resident)
#define TPB      512        // 16 waves of 32
#define RB_PER_WG 2         // 16-row blocks per WG  (256*2*16 = 8192 rows)
#define KSLICES   8         // waves splitting the K dimension per row block
#define CHUNKS    32        // 256 k-chunks of 32 / KSLICES

// d_out layout (floats): prediction | target | prediction_augment | target_augment
#define OFF_PRED    0
#define OFF_TGT     64000
#define OFF_AUG     128000
#define OFF_TGTAUG  256000

// ---------------------------------------------------------------------------
// Device-wide sense barrier (monotonic counter, zeroed by setup kernel).
// ---------------------------------------------------------------------------
__device__ inline void gbar(unsigned* cnt, unsigned target) {
  __threadfence();                       // release prior global writes (agent scope)
  __syncthreads();
  if (threadIdx.x == 0) {
    __hip_atomic_fetch_add(cnt, 1u, __ATOMIC_ACQ_REL, __HIP_MEMORY_SCOPE_AGENT);
    while (__hip_atomic_load(cnt, __ATOMIC_ACQUIRE, __HIP_MEMORY_SCOPE_AGENT) < target) {
      __builtin_amdgcn_s_sleep(1);
    }
  }
  __syncthreads();
  __threadfence();                       // acquire: invalidate stale cached lines
}

// ---------------------------------------------------------------------------
// Setup: transpose W_out (64xR -> Rx64), copy targets into d_out,
// zero bf16 state buffers and the barrier counter.
// ---------------------------------------------------------------------------
__global__ void esn_setup(const float* __restrict__ x, const float* __restrict__ Wout,
                          float* __restrict__ out, float* __restrict__ WoT,
                          __bf16* __restrict__ hbf, unsigned* __restrict__ cnt) {
  int gid = blockIdx.x * blockDim.x + threadIdx.x;
  if (gid < R * DIM) {                       // W_out^T for coalesced column access
    int j = gid >> 6, k = gid & 63;
    WoT[gid] = Wout[(size_t)k * R + j];
  }
  if (gid < STEPS * DIM) {                   // target_augment = input_sequence
    out[OFF_TGTAUG + gid] = x[gid];
    if (gid < DYN * DIM)                     // target = input_sequence[1000:]
      out[OFF_TGT + gid] = x[DYN * DIM + gid];
  }
  if (gid < 2 * R) hbf[gid] = (__bf16)0.0f;  // h0 = 0 (both ping-pong buffers)
  if (gid == 0) *cnt = 0u;
}

// ---------------------------------------------------------------------------
// Pack W_h (fp32 row-major) -> bf16 WMMA A tiles, k contiguous per lane half (same k order as B).
// Tile (rb,kc) = rows [16rb,16rb+16) x cols [32kc,32kc+32), 1KB, lane-major:
//   lane<16 : M=lane,    K = e       (e = half index 0..15)
//   lane>=16: M=lane-16, K = e+16
// ---------------------------------------------------------------------------
__global__ void esn_pack(const float* __restrict__ Wh, __bf16* __restrict__ Whp) {
  int gid  = blockIdx.x * blockDim.x + threadIdx.x;   // 512*256*32 threads
  int lane = gid & 31;
  int tile = gid >> 5;
  int kc   = tile & 255;
  int rb   = tile >> 8;
  int M    = lane & 15;
  const float* src = Wh + (size_t)(rb * 16 + M) * R + kc * 32;
  v16bf v;
#pragma unroll
  for (int e = 0; e < 16; ++e) {
    int K = (lane < 16) ? e : e + 16;
    v[e] = (__bf16)src[K];
  }
  ((v16bf*)Whp)[gid] = v;                              // 32B/lane, coalesced
}

// ---------------------------------------------------------------------------
// Persistent recurrence kernel: all 2000 steps.
//   WG wg owns rows [32wg, 32wg+32) (two 16-row WMMA blocks).
//   Wave w: row block w>>3, K-slice w&7 (32 chunks of K=32 each).
//   Step s: B1 barrier; WG0 reduces O_s partials -> Ofin, stores outputs;
//   all waves run the WMMA matvec over h_s; (pred: B2 barrier so u=Ofin is
//   visible); wave 0 finishes h_{s+1}=tanh(W_in@u + W_h@h_s), writes bf16
//   state, and emits this WG's 64-wide partial of O_{s+1}=W_out@aug(h_{s+1}).
// ---------------------------------------------------------------------------
__global__ void __launch_bounds__(TPB, 1)
esn_persistent(const __bf16* __restrict__ Whp, const float* __restrict__ WoT,
               const float* __restrict__ Win,  const float* __restrict__ x,
               float* __restrict__ out, __bf16* __restrict__ hbf,
               float* __restrict__ Opart, float* __restrict__ Ofin,
               unsigned* __restrict__ cnt) {
  __shared__ float lds_part[RB_PER_WG][KSLICES][16];
  __shared__ float lds_red[TPB];

  const int wg    = blockIdx.x;
  const int tid   = threadIdx.x;
  const int wave  = tid >> 5;
  const int lane  = tid & 31;
  const int rbl   = wave >> 3;          // 0..1
  const int slice = wave & 7;           // 0..7
  const int rb    = wg * RB_PER_WG + rbl;

  // This wave's first A tile (v16bf units): tile index rb*256 + slice*32
  const v16bf* Abase = (const v16bf*)Whp + ((size_t)rb * 256 + slice * CHUNKS) * 32 + lane;
  const int kbase = slice * CHUNKS * 32;      // first h element of this K slice
  const v16bf bzero = {};

  unsigned epoch = 0;

  for (int s = 0; s < STEPS; ++s) {
    const __bf16* hc = hbf + (s & 1) * R;          // h_s (bf16)
    __bf16*       hn = hbf + ((s + 1) & 1) * R;    // h_{s+1}

    gbar(cnt, ++epoch * NWG);   // B1: h_s and O_s partials complete everywhere

    // --- phase 1 (WG0): reduce O_s = sum over WGs, store outputs -----------
    if (wg == 0 && s >= 1) {
      int k = tid & 63, grp = tid >> 6;            // 8 groups x 32 partials
      float p = 0.f;
      for (int w2 = grp * 32; w2 < grp * 32 + 32; ++w2) p += Opart[w2 * DIM + k];
      lds_red[tid] = p;
      __syncthreads();
      if (tid < DIM) {
        float o = 0.f;
        for (int g2 = 0; g2 < 8; ++g2) o += lds_red[g2 * DIM + tid];
        Ofin[tid] = o;                             // u for prediction steps
        if (s <= DYN) out[OFF_AUG + (s - 1) * DIM + tid] = o;   // warm outputs
        if (s >= DYN) {
          out[OFF_AUG + s * DIM + tid] = o;                     // pred, augmented
          out[OFF_PRED + (s - DYN) * DIM + tid] = o;            // prediction
        }
      }
    }

    // --- WMMA matvec: this wave's 16 rows x its K slice --------------------
    // A: 16x32 bf16 tile (1KB, two b128 loads/lane). B: h chunk in column 0.
    // Two independent accumulator chains -> no TRANS32_DEP stall between
    // consecutive WMMAs; compiler pipelines the b128 loads across iterations.
    v8f acc0 = {}, acc1 = {};
#pragma unroll 2
    for (int c = 0; c < CHUNKS; c += 2) {
      v16bf a0 = Abase[(size_t)c * 32];
      v16bf a1 = Abase[(size_t)(c + 1) * 32];
      v16bf b0 = *(const v16bf*)(hc + kbase + c * 32 + (lane >> 4) * 16);
      v16bf b1 = *(const v16bf*)(hc + kbase + (c + 1) * 32 + (lane >> 4) * 16);
      acc0 = __builtin_amdgcn_wmma_f32_16x16x32_bf16(
                 false, a0, false, b0, (short)0, acc0, false, false);
      acc1 = __builtin_amdgcn_wmma_f32_16x16x32_bf16(
                 false, a1, false, b1, (short)0, acc1, false, false);
    }
    v8f acc = acc0 + acc1;

    // Column 0 of D lives in lanes 0 (M=0..7) and 16 (M=8..15), VGPRs 0..7.
    if (lane == 0 || lane == 16) {
      float* dst = &lds_part[rbl][slice][(lane >> 4) * 8];
      dst[0] = acc[0]; dst[1] = acc[1]; dst[2] = acc[2]; dst[3] = acc[3];
      dst[4] = acc[4]; dst[5] = acc[5]; dst[6] = acc[6]; dst[7] = acc[7];
    }
    __syncthreads();

    if (s >= DYN) gbar(cnt, ++epoch * NWG);  // B2: u = O_s (Ofin) now visible

    // --- phase 2 (wave 0): finish h_{s+1} for this WG's 32 rows ------------
    if (tid < 32) {
      float sum = 0.f;
#pragma unroll
      for (int w2 = 0; w2 < KSLICES; ++w2) sum += lds_part[tid >> 4][w2][tid & 15];

      const int   row = wg * 32 + tid;
      const float* wr = Win + (size_t)row * DIM;
      const float* u  = (s < DYN) ? (x + (size_t)s * DIM) : Ofin;
      float wi = 0.f;
#pragma unroll 8
      for (int j = 0; j < DIM; ++j) wi += wr[j] * u[j];

      float hv = tanhf(sum + wi);
      hn[row] = (__bf16)hv;                        // bf16 state feeds next WMMA
      float g = ((row & 1) == 0) ? hv * hv : hv;   // aug(): square even rows

      // Partial O_{s+1}[k] += sum_j Wout^T[row_j][k] * g_j over our 32 rows.
      float a0 = 0.f, a1 = 0.f;
      for (int j = 0; j < 32; ++j) {
        float gj = __shfl(g, j, 32);
        const float* wo = WoT + (size_t)(wg * 32 + j) * DIM;
        a0 += wo[tid]      * gj;
        a1 += wo[tid + 32] * gj;
      }
      Opart[wg * DIM + tid]      = a0;
      Opart[wg * DIM + tid + 32] = a1;
    }
  }
}

// ---------------------------------------------------------------------------
extern "C" void kernel_launch(void* const* d_in, const int* in_sizes, int n_in,
                              void* d_out, int out_size, void* d_ws, size_t ws_size,
                              hipStream_t stream) {
  const float* x    = (const float*)d_in[0];   // (2000, 64)
  const float* Win  = (const float*)d_in[1];   // (8192, 64)
  const float* Wh   = (const float*)d_in[2];   // (8192, 8192)
  const float* Wout = (const float*)d_in[3];   // (64, 8192)
  float* out = (float*)d_out;

  char* ws = (char*)d_ws;
  size_t off = 0;
  __bf16* Whp  = (__bf16*)(ws + off); off += (size_t)R * R * 2;          // 128 MB packed W_h
  float*  WoT  = (float*) (ws + off); off += (size_t)R * DIM * 4;        // 2 MB  W_out^T
  __bf16* hbf  = (__bf16*)(ws + off); off += (size_t)2 * R * 2;          // ping-pong bf16 state
  float*  Opart= (float*) (ws + off); off += (size_t)NWG * DIM * 4;      // per-WG output partials
  float*  Ofin = (float*) (ws + off); off += 256;                        // reduced output (u)
  unsigned* cnt= (unsigned*)(ws + off); off += 256;                      // barrier counter
  if (ws_size < off) return;   // workspace too small: refuse to launch

  esn_setup<<<(R * DIM + 255) / 256, 256, 0, stream>>>(x, Wout, out, WoT, hbf, cnt);
  esn_pack <<<(512 * 256 * 32) / 256, 256, 0, stream>>>(Wh, Whp);
  esn_persistent<<<NWG, TPB, 0, stream>>>(Whp, WoT, Win, x, out, hbf, Opart, Ofin, cnt);
}
